// ShuffleWindowAttention_41901700940108
// MI455X (gfx1250) — matrix operand order repaired and checked by reading, not compiled, not run on hardware
//
#include <hip/hip_runtime.h>

// ---- types ----
typedef __attribute__((ext_vector_type(16))) __bf16         v16bf;
typedef __attribute__((ext_vector_type(16))) unsigned short v16u;
typedef __attribute__((ext_vector_type(8)))  unsigned short v8u;
typedef __attribute__((ext_vector_type(8)))  float          v8f;

// Problem constants: B=4, N=50176, C=512, WS=49, NH=8, hd=64, w=1024
#define NTOK_N   50176
#define NWIN     1024
#define CDIM     512
#define QKVDIM   1536

// Native bf16 converts (gfx1250 hardware BF16 cvt ops)
__device__ __forceinline__ unsigned short f2bf(float f) {
  return __builtin_bit_cast(unsigned short, (__bf16)f);
}
__device__ __forceinline__ float bf2f(unsigned short h) {
  return (float)__builtin_bit_cast(__bf16, h);
}
__device__ __forceinline__ v8f zero8() {
  v8f v;
#pragma unroll
  for (int i = 0; i < 8; ++i) v[i] = 0.0f;
  return v;
}
__device__ __forceinline__ v8f wmma_bf16(v16u a, v16u b, v8f c) {
  return __builtin_amdgcn_wmma_f32_16x16x32_bf16(
      false, __builtin_bit_cast(v16bf, a),
      false, __builtin_bit_cast(v16bf, b),
      (short)0, c, false, false);
}

// A-tile 16x32 bf16 (ISA layout): lane r=lane&15 holds row r; kb=(lane>>4)*8.
// elements 0..7 = K kb+0..7, elements 8..15 = K kb+16..kb+23.
__device__ __forceinline__ v16u load_a_lds(const unsigned short* base, int stride, int lane) {
  int r = lane & 15, kb = (lane >> 4) << 3;
  const unsigned short* p = base + r * stride + kb;
  v8u lo = *(const v8u*)p;           // ds_load_b128
  v8u hi = *(const v8u*)(p + 16);    // ds_load_b128
  return __builtin_shufflevector(lo, hi, 0,1,2,3,4,5,6,7,8,9,10,11,12,13,14,15);
}
// B-tile 32x16 (KxN) where B[k][n] = M[n0+n][k0+k], M row-major with leading dim ldw:
// lane c=lane&15 -> column n0+c ; kb=(lane>>4)*8 ; elements 0..7 = K kb+0..7, 8..15 = K kb+16..23.
__device__ __forceinline__ v16u load_b_rowmajor(const unsigned short* M, int ldw,
                                                int n0, int k0, int lane) {
  int c = lane & 15, kb = (lane >> 4) << 3;
  const unsigned short* p = M + (size_t)(n0 + c) * ldw + k0 + kb;
  v8u lo = *(const v8u*)p;
  v8u hi = *(const v8u*)(p + 16);
  return __builtin_shufflevector(lo, hi, 0,1,2,3,4,5,6,7,8,9,10,11,12,13,14,15);
}

// ---- prep: fp32 weights -> bf16 in workspace ----
__global__ void swa_prep_weights(const float* __restrict__ wq, const float* __restrict__ wp,
                                 unsigned short* __restrict__ wq_bf,
                                 unsigned short* __restrict__ wp_bf) {
  int i = blockIdx.x * 256 + threadIdx.x;
  if (i < QKVDIM * CDIM) wq_bf[i] = f2bf(wq[i]);
  if (i < CDIM * CDIM)   wp_bf[i] = f2bf(wp[i]);
}

// ---- fused window-attention kernel: one window per workgroup, 8 waves ----
// Dynamic LDS (262144 B):
//   sA  [0 .. 64KB)    : phase1-2: X bf16 [64x512]; phase3+: per-head 8KB slots (S, then O)
//   sQK [64KB .. 192KB): Q|K bf16 [64x1024]; Q columns later overwritten by softmax P
//   sVt [192KB..256KB) : V transposed bf16 [512 feats x 64 tokens]
__global__ __launch_bounds__(256, 1)
void swa_fused_kernel(const float* __restrict__ x,
                      const float* __restrict__ b_qkv,
                      const float* __restrict__ b_proj,
                      const unsigned short* __restrict__ wq,   // [1536,512] bf16
                      const unsigned short* __restrict__ wp,   // [512,512]  bf16
                      float* __restrict__ out) {
  extern __shared__ unsigned short smem[];
  unsigned short* sA  = smem;                     // 32768 u16
  unsigned short* sQK = smem + 32768;             // 64*1024 u16
  unsigned short* sVt = smem + 32768 + 65536;     // 512*64 u16
  const int lane = threadIdx.x & 31;
  const int wave = threadIdx.x >> 5;
  const int bb = blockIdx.x >> 10;                // batch
  const int wi = blockIdx.x & 1023;               // window index (stride-1024 tokens)

  // ---- Phase 1: stage window X as bf16 [64x512], rows>=49 zero ----
  for (int t = threadIdx.x; t < 64 * 128; t += 256) {
    int row = t >> 7;
    int q4  = t & 127;
    float4 v = make_float4(0.f, 0.f, 0.f, 0.f);
    if (row < 49)
      v = *((const float4*)(x + ((size_t)bb * NTOK_N + (size_t)row * NWIN + wi) * CDIM) + q4);
    unsigned short* d = sA + row * CDIM + q4 * 4;
    d[0] = f2bf(v.x); d[1] = f2bf(v.y); d[2] = f2bf(v.z); d[3] = f2bf(v.w);
  }
  __syncthreads();

  // ---- Phase 2: [Q|K|V] = X @ Wqkv^T + b, 4 N-tiles per pass (A loaded once) ----
  // grp 0: nt = wave..wave+24        -> all Q columns (scaled)
  // grp 1: nt = wave+32..wave+56     -> all K columns
  // grp 2: nt = wave+64..wave+88     -> all V columns (stored transposed)
#pragma unroll 1
  for (int grp = 0; grp < 3; ++grp) {
    int ntb = wave + grp * 32;
    if (grp < 2)   // prefetch next group's weight rows (global_prefetch_b8)
      __builtin_prefetch(wq + ((size_t)((ntb + 32) * 16 + (lane & 15))) * CDIM +
                             ((lane >> 4) << 8), 0, 3);
    v8f acc[4][4];   // [j = N-tile][m = M-tile]
#pragma unroll
    for (int j = 0; j < 4; ++j)
#pragma unroll
      for (int m = 0; m < 4; ++m) acc[j][m] = zero8();
    for (int kk = 0; kk < 16; ++kk) {
      v16u bt[4];
#pragma unroll
      for (int j = 0; j < 4; ++j)
        bt[j] = load_b_rowmajor(wq, CDIM, (ntb + 8 * j) * 16, kk * 32, lane);
#pragma unroll
      for (int m = 0; m < 4; ++m) {
        v16u at = load_a_lds(sA + m * 16 * CDIM + kk * 32, CDIM, lane);
#pragma unroll
        for (int j = 0; j < 4; ++j)
          acc[j][m] = wmma_bf16(at, bt[j], acc[j][m]);
      }
    }
    if (grp < 2) {
      float scl = (grp == 0) ? 0.125f : 1.0f;    // hd^-0.5 on Q
#pragma unroll
      for (int j = 0; j < 4; ++j) {
        int col = (ntb + 8 * j) * 16 + (lane & 15);
        float bias = b_qkv[col];
#pragma unroll
        for (int m = 0; m < 4; ++m)
#pragma unroll
          for (int i = 0; i < 8; ++i) {
            int row = m * 16 + ((lane >> 4) << 3) + i;
            sQK[row * 1024 + col] = f2bf((acc[j][m][i] + bias) * scl);
          }
      }
    } else {
      // V transposed: Vt[feat][token], packed ds_store_b128 per tile
#pragma unroll
      for (int j = 0; j < 4; ++j) {
        int col = (ntb + 8 * j) * 16 + (lane & 15);
        float bias = b_qkv[col];
        int f = col - 1024;
#pragma unroll
        for (int m = 0; m < 4; ++m) {
          v8u pk;
#pragma unroll
          for (int i = 0; i < 8; ++i) pk[i] = f2bf(acc[j][m][i] + bias);
          *(v8u*)(sVt + f * 64 + m * 16 + ((lane >> 4) << 3)) = pk;
        }
      }
    }
  }
  __syncthreads();

  // ---- Phase 3: per-head (wave h) S = Qh @ Kh^T -> bf16 slot in sA ----
  {
    int h = wave;
    v8f s[16];
#pragma unroll
    for (int i = 0; i < 16; ++i) s[i] = zero8();
#pragma unroll
    for (int kk = 0; kk < 2; ++kk) {
      v16u at[4], bt[4];
#pragma unroll
      for (int m = 0; m < 4; ++m)
        at[m] = load_a_lds(sQK + m * 16 * 1024 + h * 64 + kk * 32, 1024, lane);
#pragma unroll
      for (int n = 0; n < 4; ++n)   // B[k][n] = K[token n][feat k]: contiguous in k
        bt[n] = load_b_rowmajor(sQK, 1024, n * 16, 512 + h * 64 + kk * 32, lane);
#pragma unroll
      for (int m = 0; m < 4; ++m)
#pragma unroll
        for (int n = 0; n < 4; ++n)
          s[m * 4 + n] = wmma_bf16(at[m], bt[n], s[m * 4 + n]);
    }
    unsigned short* Sh = sA + h * 4096;           // [64x64] bf16, reuses dead X region
#pragma unroll
    for (int m = 0; m < 4; ++m)
#pragma unroll
      for (int n = 0; n < 4; ++n)
#pragma unroll
        for (int i = 0; i < 8; ++i) {
          int row = m * 16 + ((lane >> 4) << 3) + i;
          int col = n * 16 + (lane & 15);
          Sh[row * 64 + col] = f2bf(s[m * 4 + n][i]);
        }
  }
  // phases 3-5 touch only this wave's head data; DS ops per wave are in-order.

  // ---- Phase 4: softmax rows of S -> P bf16 overwriting dead Q columns ----
  {
    int h = wave;
    const unsigned short* Sh = sA + h * 4096;
#pragma unroll 1
    for (int r = lane; r < 49; r += 32) {
      float vals[49];
      float mx = -1e30f;
#pragma unroll
      for (int c = 0; c < 49; ++c) { vals[c] = bf2f(Sh[r * 64 + c]); mx = fmaxf(mx, vals[c]); }
      float sum = 0.f;
#pragma unroll
      for (int c = 0; c < 49; ++c) { float e = __expf(vals[c] - mx); vals[c] = e; sum += e; }
      float inv = 1.0f / sum;
      unsigned short* Pr = sQK + r * 1024 + h * 64;
#pragma unroll
      for (int c = 0; c < 49; ++c) Pr[c] = f2bf(vals[c] * inv);
#pragma unroll
      for (int c = 49; c < 64; ++c) Pr[c] = 0;    // mask padded K tokens
    }
    for (int r = 49 + lane; r < 64; r += 32) {    // clean padded query rows
      unsigned short* Pr = sQK + r * 1024 + h * 64;
#pragma unroll
      for (int c = 0; c < 64; ++c) Pr[c] = 0;
    }
  }

  // ---- Phase 5: O = P @ V per head -> bf16 slot in sA (over dead S) ----
  {
    int h = wave;
    v8f o[16];
#pragma unroll
    for (int i = 0; i < 16; ++i) o[i] = zero8();
#pragma unroll
    for (int kk = 0; kk < 2; ++kk) {
      v16u at[4], bt[4];
#pragma unroll
      for (int m = 0; m < 4; ++m)   // A = P (stored in Q columns)
        at[m] = load_a_lds(sQK + m * 16 * 1024 + h * 64 + kk * 32, 1024, lane);
#pragma unroll
      for (int n = 0; n < 4; ++n)   // B[k][n] = V[k][feat n] = Vt[feat n][k]: contiguous
        bt[n] = load_b_rowmajor(sVt, 64, h * 64 + n * 16, kk * 32, lane);
#pragma unroll
      for (int m = 0; m < 4; ++m)
#pragma unroll
        for (int n = 0; n < 4; ++n)
          o[m * 4 + n] = wmma_bf16(at[m], bt[n], o[m * 4 + n]);
    }
    unsigned short* Oh = sA + h * 4096;           // [64x64] bf16 per head (over dead S)
#pragma unroll
    for (int m = 0; m < 4; ++m)
#pragma unroll
      for (int n = 0; n < 4; ++n)
#pragma unroll
        for (int i = 0; i < 8; ++i) {
          int row = m * 16 + ((lane >> 4) << 3) + i;
          int col = n * 16 + (lane & 15);
          Oh[row * 64 + col] = f2bf(o[m * 4 + n][i]);
        }
  }
  __syncthreads();   // phase 6 reads all heads' O across waves

  // ---- Phase 6: out = O @ Wp^T + b; 4 N-tiles at once; un-shuffled fp32 store ----
  {
    v8f acc[4][4];
#pragma unroll
    for (int j = 0; j < 4; ++j)
#pragma unroll
      for (int m = 0; m < 4; ++m) acc[j][m] = zero8();
    for (int kk = 0; kk < 16; ++kk) {
      v16u bt[4];
#pragma unroll
      for (int j = 0; j < 4; ++j)
        bt[j] = load_b_rowmajor(wp, CDIM, (wave + 8 * j) * 16, kk * 32, lane);
      int r = lane & 15, kb = (lane >> 4) << 3;
      int c0 = kk * 32 + kb, c1 = c0 + 16;        // K chunks: [c0,c0+8), [c1,c1+8)
#pragma unroll
      for (int m = 0; m < 4; ++m) {
        const unsigned short* p0 = sA + (c0 >> 6) * 4096 + (m * 16 + r) * 64 + (c0 & 63);
        const unsigned short* p1 = sA + (c1 >> 6) * 4096 + (m * 16 + r) * 64 + (c1 & 63);
        v8u lo = *(const v8u*)p0;
        v8u hi = *(const v8u*)p1;
        v16u at = __builtin_shufflevector(lo, hi, 0,1,2,3,4,5,6,7,8,9,10,11,12,13,14,15);
#pragma unroll
        for (int j = 0; j < 4; ++j)
          acc[j][m] = wmma_bf16(at, bt[j], acc[j][m]);
      }
    }
#pragma unroll
    for (int j = 0; j < 4; ++j) {
      int col = (wave + 8 * j) * 16 + (lane & 15);
      float bias = b_proj[col];
#pragma unroll
      for (int m = 0; m < 4; ++m)
#pragma unroll
        for (int i = 0; i < 8; ++i) {
          int srow = m * 16 + ((lane >> 4) << 3) + i;
          if (srow < 49)
            out[((size_t)bb * NTOK_N + (size_t)srow * NWIN + wi) * CDIM + col] =
                acc[j][m][i] + bias;
        }
    }
  }
}

extern "C" void kernel_launch(void* const* d_in, const int* in_sizes, int n_in,
                              void* d_out, int out_size, void* d_ws, size_t ws_size,
                              hipStream_t stream) {
  (void)in_sizes; (void)n_in; (void)out_size; (void)ws_size;
  const float* x      = (const float*)d_in[0];
  const float* w_qkv  = (const float*)d_in[1];
  const float* b_qkv  = (const float*)d_in[2];
  const float* w_proj = (const float*)d_in[3];
  const float* b_proj = (const float*)d_in[4];
  float* out = (float*)d_out;

  unsigned short* wq_bf = (unsigned short*)d_ws;                              // 1536*512
  unsigned short* wp_bf = (unsigned short*)((char*)d_ws + (size_t)QKVDIM * CDIM * 2);

  // bf16 weight prep (~2 MB in d_ws)
  swa_prep_weights<<<(QKVDIM * CDIM + 255) / 256, 256, 0, stream>>>(w_qkv, w_proj, wq_bf, wp_bf);

  // 256 KB dynamic LDS per workgroup (<= 320 KB WGP LDS)
  hipFuncSetAttribute((const void*)swa_fused_kernel,
                      hipFuncAttributeMaxDynamicSharedMemorySize, 262144);
  swa_fused_kernel<<<4096, 256, 262144, stream>>>(x, b_qkv, b_proj, wq_bf, wp_bf, out);
}
